// RNN_67430986547643
// MI455X (gfx1250) — matrix operand order repaired and checked by reading, not compiled, not run on hardware
//
#include <hip/hip_runtime.h>
#include <hip/hip_bf16.h>
#include <stdint.h>

// ---------------- problem constants ----------------
#define B_SZ   512
#define N_ORN  200
#define N_IN   2000
#define NEUR   4500
#define TSTEPS 8
#define NCLS   1000

// padded sizes: P is a multiple of 64 (N-tile) and 32 (K-tile)
#define P       4544
#define KT      (P / 32)        // 142 K-tiles of 32
#define NT_W    (P / 16)        // 284 16-col tiles in W
#define NOUT_P  1024            // padded class count (mult of 64)
#define NT_O    (NOUT_P / 16)   // 64

typedef __attribute__((ext_vector_type(16))) __bf16   bf16x16;
typedef __attribute__((ext_vector_type(8)))  float    f32x8;
typedef __attribute__((ext_vector_type(4)))  uint32_t u32x4;

// global-address-space pointer to a 16B vector (forces global_load_b128)
typedef const u32x4 __attribute__((address_space(1)))* gq4p;

union FragU { u32x4 q[2]; bf16x16 v; };

__device__ __forceinline__ uint16_t f32_to_bf16(float f) {
  union { float f; uint32_t u; } c; c.f = f;
  uint32_t u = c.u;
  return (uint16_t)((u + 0x7FFFu + ((u >> 16) & 1u)) >> 16);  // RNE
}

// A fragment: two contiguous 8-elem runs at +0 and +16 elements (ISA 16-bit A layout)
__device__ __forceinline__ void loadA(FragU& f, const uint16_t* p) {
  f.q[0] = *(gq4p)(p);
  f.q[1] = *(gq4p)(p + 16);
}
// B fragment: 16 contiguous pre-packed elements (32 bytes)
__device__ __forceinline__ void loadB(FragU& f, const uint16_t* p) {
  f.q[0] = *(gq4p)(p);
  f.q[1] = *(gq4p)(p + 8);
}

// WGP-scope prefetch (scope 0 = pull into all cache levels incl. WGP$)
__device__ __forceinline__ void prefetch_wgp(const uint16_t* p) {
  asm volatile("global_prefetch_b8 %0, off" :: "v"(p));
}

__device__ __forceinline__ f32x8 wmma_bf16(const FragU& a, const FragU& b, f32x8 c) {
  return __builtin_amdgcn_wmma_f32_16x16x32_bf16(
      false, a.v, false, b.v, (short)0, c, false, false);
}

__device__ __forceinline__ void compute4(
    f32x8 acc[4], const FragU& A,
    const FragU& B0, const FragU& B1, const FragU& B2, const FragU& B3)
{
  acc[0] = wmma_bf16(A, B0, acc[0]);
  acc[1] = wmma_bf16(A, B1, acc[1]);
  acc[2] = wmma_bf16(A, B2, acc[2]);
  acc[3] = wmma_bf16(A, B3, acc[3]);
}

// ---------------------------------------------------------------------------
// Core K-loop: acc[0..3] += A-strip(16xK) x 4 B-strips(Kx16).
// Manual 2-stage software pipeline, 2 K-chunks per rolled iteration.
// sched_barrier(0) fences keep loads a full stage ahead of the WMMAs so the
// waitcnt pass emits partial waits instead of full drains.
// ---------------------------------------------------------------------------
__device__ __forceinline__ void gemm_acc(
    f32x8 acc[4],
    const uint16_t* aP,
    const uint16_t* b0, const uint16_t* b1,
    const uint16_t* b2, const uint16_t* b3)
{
  FragU Aa, B0a, B1a, B2a, B3a;   // even chunk buffers
  FragU Ab, B0b, B1b, B2b, B3b;   // odd  chunk buffers

  // preload chunk 0
  loadA(Aa, aP);
  loadB(B0a, b0); loadB(B1a, b1); loadB(B2a, b2); loadB(B3a, b3);

#pragma unroll 1
  for (int i = 0; i < (KT - 2) / 2; ++i) {      // 70 iterations, chunks 0..139
    // load chunk 2i+1; prefetch W stream ~8 chunks ahead into WGP$
    loadA(Ab, aP + 32);
    loadB(B0b, b0 + 512); loadB(B1b, b1 + 512);
    loadB(B2b, b2 + 512); loadB(B3b, b3 + 512);
    prefetch_wgp(b0 + 8 * 512); prefetch_wgp(b1 + 8 * 512);
    prefetch_wgp(b2 + 8 * 512); prefetch_wgp(b3 + 8 * 512);
    __builtin_amdgcn_sched_barrier(0);
    compute4(acc, Aa, B0a, B1a, B2a, B3a);      // chunk 2i
    __builtin_amdgcn_sched_barrier(0);
    // load chunk 2i+2
    loadA(Aa, aP + 64);
    loadB(B0a, b0 + 1024); loadB(B1a, b1 + 1024);
    loadB(B2a, b2 + 1024); loadB(B3a, b3 + 1024);
    __builtin_amdgcn_sched_barrier(0);
    compute4(acc, Ab, B0b, B1b, B2b, B3b);      // chunk 2i+1
    __builtin_amdgcn_sched_barrier(0);
    aP += 64; b0 += 1024; b1 += 1024; b2 += 1024; b3 += 1024;
  }
  // tail: a-buf holds chunk KT-2; fetch and fold the final two chunks
  loadA(Ab, aP + 32);
  loadB(B0b, b0 + 512); loadB(B1b, b1 + 512);
  loadB(B2b, b2 + 512); loadB(B3b, b3 + 512);
  compute4(acc, Aa, B0a, B1a, B2a, B3a);        // chunk KT-2
  compute4(acc, Ab, B0b, B1b, B2b, B3b);        // chunk KT-1
}

// ---------------------------------------------------------------------------
// Pack |src| (row-major [kMax x nMax] fp32) into WMMA B-fragment order, bf16.
// Packed index: (((ntile*KT + ktile)*32 + lane)*16 + e)
//   value = |src[ ktile*32 + (lane>>4)*8 + (e&7) + (e>>3)*16 ][ ntile*16 + (lane&15) ]|
// ---------------------------------------------------------------------------
__global__ __launch_bounds__(256) void pack_b_kernel(
    const float* __restrict__ src, int srcN, int kMax, int nMax,
    uint16_t* __restrict__ dst)
{
  const size_t idx  = (size_t)blockIdx.x * 256 + threadIdx.x;
  const int    e    = (int)(idx & 15u);
  const int    lane = (int)((idx >> 4) & 31u);
  const size_t blk  = idx >> 9;
  const int    kt   = (int)(blk % KT);
  const int    nt   = (int)(blk / KT);
  const int    k    = kt * 32 + (lane >> 4) * 8 + (e & 7) + (e >> 3) * 16;
  const int    n    = nt * 16 + (lane & 15);
  float v = 0.0f;
  if (k < kMax && n < nMax) v = fabsf(src[(size_t)k * srcN + n]);
  dst[idx] = f32_to_bf16(v);
}

// h0[b][c] = (c < N_IN) ? x[b][c % N_ORN] + noise[b][c] : 0   (bf16, pitch P)
__global__ __launch_bounds__(256) void init_h_kernel(
    const float* __restrict__ x, const float* __restrict__ noise,
    uint16_t* __restrict__ h)
{
  const int c = blockIdx.x * 256 + threadIdx.x;
  const int b = blockIdx.y;
  if (c >= P) return;
  float v = 0.0f;
  if (c < N_IN) v = x[b * N_ORN + (c % N_ORN)] + noise[b * N_IN + c];
  h[(size_t)b * P + c] = f32_to_bf16(v);
}

// ---------------------------------------------------------------------------
// One recurrence step: hOut = relu(hIn @ Wpacked + b_rnn), bf16 in/out.
// 256 threads = 8 waves; wave owns rows [m0,m0+16) x cols [n0,n0+64).
// ---------------------------------------------------------------------------
__global__ __launch_bounds__(256) void rnn_step_kernel(
    const uint16_t* __restrict__ hIn,
    const uint16_t* __restrict__ wPack,
    const float*    __restrict__ bias,
    uint16_t*       __restrict__ hOut)
{
  const int tid  = threadIdx.x;
  const int wave = tid >> 5;
  const int lane = tid & 31;
  const int ln   = lane & 15;
  const int g    = lane >> 4;
  const int m0   = blockIdx.x * 128 + wave * 16;
  const int n0   = blockIdx.y * 64;

  f32x8 acc[4];
#pragma unroll
  for (int t = 0; t < 4; ++t)
    acc[t] = (f32x8){0.f, 0.f, 0.f, 0.f, 0.f, 0.f, 0.f, 0.f};

  const uint16_t* aP = hIn + (size_t)(m0 + ln) * P + g * 8;
  const uint16_t* b0 = wPack + ((size_t)(n0 / 16 + 0) * KT * 32 + lane) * 16;
  const uint16_t* b1 = wPack + ((size_t)(n0 / 16 + 1) * KT * 32 + lane) * 16;
  const uint16_t* b2 = wPack + ((size_t)(n0 / 16 + 2) * KT * 32 + lane) * 16;
  const uint16_t* b3 = wPack + ((size_t)(n0 / 16 + 3) * KT * 32 + lane) * 16;

  gemm_acc(acc, aP, b0, b1, b2, b3);

  // D layout -> row m = m0 + g*8 + r, col n = n0 + t*16 + ln
#pragma unroll
  for (int t = 0; t < 4; ++t) {
    const int   n = n0 + t * 16 + ln;
    const float b = (n < NEUR) ? bias[n] : 0.0f;
#pragma unroll
    for (int r = 0; r < 8; ++r) {
      const int m = m0 + g * 8 + r;
      float v = acc[t][r] + b;
      v = fmaxf(v, 0.0f);
      hOut[(size_t)m * P + n] = f32_to_bf16(v);
    }
  }
}

// ---------------------------------------------------------------------------
// Readout: logits = h @ |w_out| + b_out  (f32 output, N limited to 1000)
// ---------------------------------------------------------------------------
__global__ __launch_bounds__(256) void out_gemm_kernel(
    const uint16_t* __restrict__ hIn,
    const uint16_t* __restrict__ oPack,
    const float*    __restrict__ b_out,
    float*          __restrict__ out)
{
  const int tid  = threadIdx.x;
  const int wave = tid >> 5;
  const int lane = tid & 31;
  const int ln   = lane & 15;
  const int g    = lane >> 4;
  const int m0   = blockIdx.x * 128 + wave * 16;
  const int n0   = blockIdx.y * 64;

  f32x8 acc[4];
#pragma unroll
  for (int t = 0; t < 4; ++t)
    acc[t] = (f32x8){0.f, 0.f, 0.f, 0.f, 0.f, 0.f, 0.f, 0.f};

  const uint16_t* aP = hIn + (size_t)(m0 + ln) * P + g * 8;
  const uint16_t* b0 = oPack + ((size_t)(n0 / 16 + 0) * KT * 32 + lane) * 16;
  const uint16_t* b1 = oPack + ((size_t)(n0 / 16 + 1) * KT * 32 + lane) * 16;
  const uint16_t* b2 = oPack + ((size_t)(n0 / 16 + 2) * KT * 32 + lane) * 16;
  const uint16_t* b3 = oPack + ((size_t)(n0 / 16 + 3) * KT * 32 + lane) * 16;

  gemm_acc(acc, aP, b0, b1, b2, b3);

#pragma unroll
  for (int t = 0; t < 4; ++t) {
    const int n = n0 + t * 16 + ln;
    if (n < NCLS) {
      const float b = b_out[n];
#pragma unroll
      for (int r = 0; r < 8; ++r) {
        const int m = m0 + g * 8 + r;
        out[(size_t)m * NCLS + n] = acc[t][r] + b;
      }
    }
  }
}

// ---------------------------------------------------------------------------
extern "C" void kernel_launch(void* const* d_in, const int* in_sizes, int n_in,
                              void* d_out, int out_size, void* d_ws, size_t ws_size,
                              hipStream_t stream) {
  const float* x     = (const float*)d_in[0];
  const float* noise = (const float*)d_in[1];
  const float* w_rnn = (const float*)d_in[2];
  const float* b_rnn = (const float*)d_in[3];
  const float* w_out = (const float*)d_in[4];
  const float* b_out = (const float*)d_in[5];
  float* out = (float*)d_out;

  char* ws = (char*)d_ws;
  size_t off = 0;
  uint16_t* wPack = (uint16_t*)(ws + off); off += (size_t)P * P * 2;        // 41.3 MB
  uint16_t* oPack = (uint16_t*)(ws + off); off += (size_t)P * NOUT_P * 2;   //  9.3 MB
  uint16_t* hA    = (uint16_t*)(ws + off); off += (size_t)B_SZ * P * 2;     //  4.65 MB
  uint16_t* hB    = (uint16_t*)(ws + off); off += (size_t)B_SZ * P * 2;     //  4.65 MB
  (void)ws_size; (void)in_sizes; (void)n_in; (void)out_size;

  // one-time prep: |W| -> bf16 B-fragment layout; |w_out| likewise; h0
  {
    const size_t tot = (size_t)NT_W * KT * 512;            // = P*P
    pack_b_kernel<<<(unsigned)(tot / 256), 256, 0, stream>>>(
        w_rnn, NEUR, NEUR, NEUR, wPack);
  }
  {
    const size_t tot = (size_t)NT_O * KT * 512;            // = P*NOUT_P
    pack_b_kernel<<<(unsigned)(tot / 256), 256, 0, stream>>>(
        w_out, NCLS, NEUR, NCLS, oPack);
  }
  init_h_kernel<<<dim3((P + 255) / 256, B_SZ), 256, 0, stream>>>(x, noise, hA);

  // 8 sequential recurrence steps (inter-step sync = kernel boundary)
  uint16_t* src = hA;
  uint16_t* dst = hB;
  for (int t = 0; t < TSTEPS; ++t) {
    rnn_step_kernel<<<dim3(B_SZ / 128, P / 64), 256, 0, stream>>>(
        src, wPack, b_rnn, dst);
    uint16_t* tmp = src; src = dst; dst = tmp;
  }

  // readout GEMM, f32 output
  out_gemm_kernel<<<dim3(B_SZ / 128, NOUT_P / 64), 256, 0, stream>>>(
      src, oPack, b_out, out);
}
